// Upsampler_52424370815577
// MI455X (gfx1250) — matrix operand order repaired and checked by reading, not compiled, not run on hardware
//
#include <hip/hip_runtime.h>
#include <hip/hip_bf16.h>
#include <math.h>

typedef __attribute__((ext_vector_type(16))) _Float16 v16h;
typedef __attribute__((ext_vector_type(8)))  float    v8f;

constexpr int Bn = 2, Hc = 256, Wc = 256, HWc = Hc * Wc, NPIX = Bn * HWc;
constexpr int H2 = 512, W2 = 512;  constexpr int S2 = H2 * W2;
constexpr int PH = Hc + 4, PW = Wc + 4;

static __device__ __forceinline__ int clampi(int v, int lo, int hi) {
  return v < lo ? lo : (v > hi ? hi : v);
}
static __device__ __forceinline__ float sigm(float x) { return 1.f / (1.f + expf(-x)); }
static __device__ __forceinline__ void cubw(float t, float* w) {
  // Catmull-Rom (a=-0.5), taps at floor-1..floor+2
  float t2 = t * t, t3 = t2 * t;
  w[0] = -0.5f * t3 + t2 - 0.5f * t;
  w[1] =  1.5f * t3 - 2.5f * t2 + 1.f;
  w[2] = -1.5f * t3 + 2.f * t2 + 0.5f * t;
  w[3] =  0.5f * t3 - 0.5f * t2;
}

// ---------------- weight / bias packing (f32 -> padded f16 [CoPad][Kpad]) ---------
__global__ void k_pack_weight(const float* __restrict__ w, _Float16* __restrict__ out,
                              int Co, int Ci, int CoPad, int Kpad) {
  int idx = blockIdx.x * blockDim.x + threadIdx.x;
  if (idx >= CoPad * Kpad) return;
  int o = idx / Kpad, k = idx - o * Kpad;
  float v = (o < Co && k < Ci) ? w[o * Ci + k] : 0.f;
  out[idx] = (_Float16)v;
}
__global__ void k_pack_bias(const float* __restrict__ b, float* __restrict__ out,
                            int Co, int CoPad) {
  int idx = blockIdx.x * blockDim.x + threadIdx.x;
  if (idx >= CoPad) return;
  out[idx] = (idx < Co) ? b[idx] : 0.f;
}

// ---------------- build ctx_all activations (290ch -> 320 pad), pe-MLP inline ----
__global__ void k_pack_act(const float* __restrict__ flow, const float* __restrict__ cert,
                           const float* __restrict__ feat, const float* __restrict__ ctx,
                           const float* __restrict__ w1, const float* __restrict__ b1,
                           const float* __restrict__ w2, const float* __restrict__ b2,
                           const float* __restrict__ w3, const float* __restrict__ b3,
                           _Float16* __restrict__ act) {
  __shared__ float sp[1086];  // w1[64] b1[16] w2[512] b2[32] w3[448] b3[14]
  for (int t = threadIdx.x; t < 1086; t += blockDim.x) {
    float v;
    if      (t < 64)   v = w1[t];
    else if (t < 80)   v = b1[t - 64];
    else if (t < 592)  v = w2[t - 80];
    else if (t < 624)  v = b2[t - 592];
    else if (t < 1072) v = w3[t - 624];
    else               v = b3[t - 1072];
    sp[t] = v;
  }
  __syncthreads();
  int p = blockIdx.x * blockDim.x + threadIdx.x;
  if (p >= NPIX) return;
  int b = p / HWc, hw = p - b * HWc, i = hw / Wc, j = hw - i * Wc;
  float f0 = flow[(size_t)(b * 2 + 0) * HWc + hw];
  float f1 = flow[(size_t)(b * 2 + 1) * HWc + hw];
  float x0 = f0 + 1.f, x1 = f1 + 1.f;
  float x2 = 2.f * (float)j / (float)(Wc - 1);   // gx + 1
  float x3 = 2.f * (float)i / (float)(Hc - 1);   // gy + 1
  float h1[16];
  for (int o = 0; o < 16; ++o) {
    const float* wr = &sp[o * 4];
    h1[o] = fmaxf(wr[0] * x0 + wr[1] * x1 + wr[2] * x2 + wr[3] * x3 + sp[64 + o], 0.f);
  }
  float h2[32];
  for (int o = 0; o < 32; ++o) {
    const float* wr = &sp[80 + o * 16];
    float v = sp[592 + o];
    for (int c = 0; c < 16; ++c) v += wr[c] * h1[c];
    h2[o] = fmaxf(v, 0.f);
  }
  float h3[14];
  for (int o = 0; o < 14; ++o) {
    const float* wr = &sp[624 + o * 32];
    float v = sp[1072 + o];
    for (int c = 0; c < 32; ++c) v += wr[c] * h2[c];
    h3[o] = v;
  }
  _Float16* row = act + (size_t)p * 320;
  for (int c = 0; c < 128; ++c) row[c]       = (_Float16)feat[(size_t)(b * 128 + c) * HWc + hw];
  for (int c = 0; c < 144; ++c) row[128 + c] = (_Float16)ctx [(size_t)(b * 144 + c) * HWc + hw];
  row[272] = (_Float16)tanhf(cert[(size_t)(b * 2 + 0) * HWc + hw] * 0.1f);
  row[273] = (_Float16)tanhf(cert[(size_t)(b * 2 + 1) * HWc + hw] * 0.1f);
  for (int c = 0; c < 14; ++c) row[274 + c] = (_Float16)h3[c];
  row[288] = (_Float16)f0;  row[289] = (_Float16)f1;
  for (int c = 290; c < 320; ++c) row[c] = (_Float16)0.f;
}

// ---------------- WMMA GEMM: out[p][n] = act[p][:] . wt[n][:] + bias[n] ----------
// mode 0: f16 [p][outStride]; mode 1: f32 [p][outStride]; mode 2: f32 NCHW (Cn chans)
__global__ void k_gemm(const _Float16* __restrict__ act, const _Float16* __restrict__ wt,
                       const float* __restrict__ bias, void* __restrict__ outv,
                       int K, int nTiles, int outStride, int relu, int mode, int Cn) {
  int nTile = blockIdx.y * blockDim.y + threadIdx.y;
  if (nTile >= nTiles) return;
  int lane = threadIdx.x;
  int half = lane >> 4, l16 = lane & 15;
  int pixBase = blockIdx.x * 16;
  int nBase = nTile * 16;
  const _Float16* aRow = act + (size_t)(pixBase + l16) * K + half * 8;
  const _Float16* bRow = wt  + (size_t)(nBase  + l16) * K + half * 8;
  v8f c = {};
  for (int k0 = 0; k0 < K; k0 += 32) {
    v16h a, b;
    ((float4*)&a)[0] = *(const float4*)(aRow + k0);        // K = k0+half*8   .. +7
    ((float4*)&a)[1] = *(const float4*)(aRow + k0 + 16);   // K = k0+16+half*8.. +7
    ((float4*)&b)[0] = *(const float4*)(bRow + k0);        // K = k0+half*8   .. +7
    ((float4*)&b)[1] = *(const float4*)(bRow + k0 + 16);
    __builtin_prefetch(aRow + k0 + 64, 0, 1);
    c = __builtin_amdgcn_wmma_f32_16x16x32_f16(false, a, false, b, (short)0, c, false, false);
  }
  int n = nBase + l16;
  float bv = bias[n];
  for (int r = 0; r < 8; ++r) {
    int p = pixBase + r + 8 * half;
    float v = c[r] + bv;
    if (relu) v = fmaxf(v, 0.f);
    if (mode == 0)      ((_Float16*)outv)[(size_t)p * outStride + n] = (_Float16)v;
    else if (mode == 1) ((float*)outv)[(size_t)p * outStride + n] = v;
    else {
      int bb = p / HWc, hw = p - bb * HWc;
      ((float*)outv)[((size_t)bb * Cn + n) * HWc + hw] = v;
    }
  }
}

// ---------------- q/k normalize into padded [B][H+4][W+4][32] (-1 border) --------
__global__ void k_qk_prepare(const float* __restrict__ qk,
                             float* __restrict__ qp, float* __restrict__ kp) {
  int idx = blockIdx.x * blockDim.x + threadIdx.x;
  if (idx >= Bn * PH * PW) return;
  int b = idx / (PH * PW), r = idx - b * PH * PW, ip = r / PW, jp = r - ip * PW;
  float* qo = qp + (size_t)idx * 32;
  float* ko = kp + (size_t)idx * 32;
  if (ip < 2 || ip >= Hc + 2 || jp < 2 || jp >= Wc + 2) {
    for (int c = 0; c < 32; ++c) { qo[c] = -1.f; ko[c] = -1.f; }
    return;
  }
  int i = ip - 2, j = jp - 2;
  const float* row = qk + ((size_t)b * HWc + i * Wc + j) * 64;
  float q[32], k[32], sq = 0.f, sk = 0.f;
  for (int c = 0; c < 32; ++c) { q[c] = row[c];      sq += q[c] * q[c]; }
  for (int c = 0; c < 32; ++c) { k[c] = row[32 + c]; sk += k[c] * k[c]; }
  float rq = rsqrtf(sq + 1e-12f), rk = rsqrtf(sk + 1e-12f);
  for (int c = 0; c < 32; ++c) { qo[c] = q[c] * rq; ko[c] = k[c] * rk; }
}

// ---------------- ctx_new = [f[:,16:160] (144), cv (25), pad (23)] f16 -----------
__global__ void k_ctx_new(const _Float16* __restrict__ fout, const float* __restrict__ qp,
                          const float* __restrict__ kp, _Float16* __restrict__ ctxNew) {
  int p = blockIdx.x * blockDim.x + threadIdx.x;
  if (p >= NPIX) return;
  int b = p / HWc, hw = p - b * HWc, i = hw / Wc, j = hw - i * Wc;
  _Float16* row = ctxNew + (size_t)p * 192;
  const _Float16* fr = fout + (size_t)p * 160 + 16;   // f[:, -(HID+LOC):]
  for (int c = 0; c < 144; ++c) row[c] = fr[c];
  const float* qc = qp + ((size_t)(b * PH + i + 2) * PW + (j + 2)) * 32;
  float q[32];
  for (int c = 0; c < 32; ++c) q[c] = qc[c];
  for (int di = 0; di < 5; ++di)
    for (int dj = 0; dj < 5; ++dj) {
      size_t base = ((size_t)(b * PH + i + di) * PW + (j + dj)) * 32;
      float s = 0.f;
      for (int c = 0; c < 32; ++c) s += q[c] * (kp[base + c] + qp[base + c]);
      row[144 + di * 5 + dj] = (_Float16)s;
    }
  for (int c = 169; c < 192; ++c) row[c] = (_Float16)0.f;
}

// ---------------- fus input = [cnew (128), context (144), pad (16)] f16 ----------
__global__ void k_fus_in(const _Float16* __restrict__ cnew, const float* __restrict__ ctx,
                         _Float16* __restrict__ fusIn) {
  int p = blockIdx.x * blockDim.x + threadIdx.x;
  if (p >= NPIX) return;
  int b = p / HWc, hw = p - b * HWc;
  _Float16* row = fusIn + (size_t)p * 288;
  const _Float16* cr = cnew + (size_t)p * 128;
  for (int c = 0; c < 128; ++c) row[c] = cr[c];
  for (int c = 0; c < 144; ++c) row[128 + c] = (_Float16)ctx[(size_t)(b * 144 + c) * HWc + hw];
  for (int c = 272; c < 288; ++c) row[c] = (_Float16)0.f;
}

// ---------------- batch-norm statistics (atomics into 16-float scratch) ----------
__global__ void k_zero_stats(float* __restrict__ stats) {
  if (threadIdx.x < 16) stats[threadIdx.x] = 0.f;
}
__global__ void k_score_stats(const float* __restrict__ s, float* __restrict__ stats) {
  __shared__ float acc[8];
  if (threadIdx.x < 8) acc[threadIdx.x] = 0.f;
  __syncthreads();
  int p = blockIdx.x * blockDim.x + threadIdx.x;
  if (p < NPIX) {
    const float* row = s + (size_t)p * 48;
    for (int g = 0; g < 4; ++g) {
      float sm = 0.f, sq = 0.f;
      for (int n = 0; n < 9; ++n) { float v = row[g * 9 + n]; sm += v; sq += v * v; }
      atomicAdd(&acc[g], sm);
      atomicAdd(&acc[4 + g], sq);
    }
  }
  __syncthreads();
  if (threadIdx.x < 8) atomicAdd(&stats[threadIdx.x], acc[threadIdx.x]);
}
__global__ void k_cd_stats(const float* __restrict__ ch, float* __restrict__ stats) {
  __shared__ float acc[4];
  if (threadIdx.x < 4) acc[threadIdx.x] = 0.f;
  __syncthreads();
  int idx = blockIdx.x * blockDim.x + threadIdx.x;
  if (idx < Bn * 2 * S2) {
    int c = (idx / S2) % 2;
    float v = ch[idx];
    atomicAdd(&acc[c], v);
    atomicAdd(&acc[2 + c], v * v);
  }
  __syncthreads();
  if (threadIdx.x < 4) atomicAdd(&stats[8 + threadIdx.x], acc[threadIdx.x]);
}

// ---------------- convex upsample + bicubic blend -> flow_out --------------------
__global__ void k_upsample(const float* __restrict__ s, const float* __restrict__ flow,
                           const float* __restrict__ stats, float* __restrict__ outFlow) {
  int p = blockIdx.x * blockDim.x + threadIdx.x;
  if (p >= NPIX) return;
  int b = p / HWc, hw = p - b * HWc, i = hw / Wc, j = hw - i * Wc;
  const float cnt = (float)(Bn * 9 * HWc);
  const float* row = s + (size_t)p * 48;
  float sm[36];
  for (int g = 0; g < 4; ++g) {
    float m = stats[g] / cnt;
    float var = stats[4 + g] / cnt - m * m;
    float rstd = rsqrtf(var + 1e-5f);
    float v[9], mx = -1e30f;
    for (int n = 0; n < 9; ++n) { v[n] = (row[g * 9 + n] - m) * rstd; mx = fmaxf(mx, v[n]); }
    float ssum = 0.f;
    for (int n = 0; n < 9; ++n) { v[n] = expf(v[n] - mx); ssum += v[n]; }
    float inv = 1.f / ssum;
    for (int n = 0; n < 9; ++n) sm[g * 9 + n] = v[n] * inv;
  }
  float nb[2][9];
  for (int cch = 0; cch < 2; ++cch) {
    const float* fc = flow + (size_t)(b * 2 + cch) * HWc;
    for (int di = 0; di < 3; ++di)
      for (int dj = 0; dj < 3; ++dj) {
        int ii = i + di - 1, jj = j + dj - 1;
        nb[cch][di * 3 + dj] =
            (ii >= 0 && ii < Hc && jj >= 0 && jj < Wc) ? fc[ii * Wc + jj] : 0.f;
      }
  }
  float up[2][4];
  for (int cch = 0; cch < 2; ++cch)
    for (int g = 0; g < 4; ++g) {
      float a = 0.f;
      for (int n = 0; n < 9; ++n) a += sm[g * 9 + n] * nb[cch][n];
      up[cch][g] = a;
    }
  for (int a = 0; a < 2; ++a) {
    float sy = (float)i + (a ? 0.25f : -0.25f);
    int iyf = (int)floorf(sy);
    float wy[4]; cubw(sy - (float)iyf, wy);
    for (int bb = 0; bb < 2; ++bb) {
      float sx = (float)j + (bb ? 0.25f : -0.25f);
      int ixf = (int)floorf(sx);
      float wx[4]; cubw(sx - (float)ixf, wx);
      float alpha = row[36 + a * 2 + bb] * 0.1f;
      float sA = sigm(alpha);
      int Oy = 2 * i + a, Ox = 2 * j + bb;
      for (int cch = 0; cch < 2; ++cch) {
        const float* fc = flow + (size_t)(b * 2 + cch) * HWc;
        float fo = 0.f;
        for (int ky = 0; ky < 4; ++ky) {
          int yy = clampi(iyf - 1 + ky, 0, Hc - 1);
          for (int kx = 0; kx < 4; ++kx) {
            int xx = clampi(ixf - 1 + kx, 0, Wc - 1);
            fo += wy[ky] * wx[kx] * fc[yy * Wc + xx];
          }
        }
        float val = up[cch][a * 2 + bb] * sA + fo * (1.f - sA);
        outFlow[((size_t)(b * 2 + cch) * H2 + Oy) * W2 + Ox] = val;
      }
    }
  }
}

// ---------------- certainty-decoder input (tanh(.../3)) + cert_old ---------------
__global__ void k_cd_in(const float* __restrict__ s, const float* __restrict__ cert,
                        float* __restrict__ cdin, float* __restrict__ certOld) {
  int idx = blockIdx.x * blockDim.x + threadIdx.x;
  if (idx >= Bn * S2) return;
  int b = idx / S2, r = idx - b * S2, Oy = r / W2, Ox = r - Oy * W2;
  int i = Oy >> 1, a = Oy & 1, j = Ox >> 1, bb = Ox & 1;
  int p = b * HWc + i * Wc + j;
  float sy = (Oy + 0.5f) * 0.5f - 0.5f;
  int y0 = (int)floorf(sy); float ty = sy - (float)y0;
  int y1 = clampi(y0 + 1, 0, Hc - 1); y0 = clampi(y0, 0, Hc - 1);
  float sx = (Ox + 0.5f) * 0.5f - 0.5f;
  int x0 = (int)floorf(sx); float tx = sx - (float)x0;
  int x1 = clampi(x0 + 1, 0, Wc - 1); x0 = clampi(x0, 0, Wc - 1);
  float co[2];
  for (int cch = 0; cch < 2; ++cch) {
    const float* cp = cert + (size_t)(b * 2 + cch) * HWc;
    float v00 = cp[y0 * Wc + x0], v01 = cp[y0 * Wc + x1];
    float v10 = cp[y1 * Wc + x0], v11 = cp[y1 * Wc + x1];
    co[cch] = (1.f - ty) * ((1.f - tx) * v00 + tx * v01) +
              ty * ((1.f - tx) * v10 + tx * v11);
    certOld[(size_t)(b * 2 + cch) * S2 + Oy * W2 + Ox] = co[cch];
  }
  int sub = a * 2 + bb;
  float cn1 = s[(size_t)p * 48 + 36 + 4 + sub];
  float cn2 = s[(size_t)p * 48 + 36 + 8 + sub];
  size_t base = (size_t)b * 4 * S2 + (size_t)Oy * W2 + Ox;
  cdin[base + 0 * (size_t)S2] = tanhf(cn1 * (1.f / 3.f));
  cdin[base + 1 * (size_t)S2] = tanhf(cn2 * (1.f / 3.f));
  cdin[base + 2 * (size_t)S2] = tanhf(co[0] * (1.f / 3.f));
  cdin[base + 3 * (size_t)S2] = tanhf(co[1] * (1.f / 3.f));
}

// ---------------- cd conv chain (small channel counts, VALU) ---------------------
__global__ void k_conv3_4_16(const float* __restrict__ in, const float* __restrict__ w,
                             const float* __restrict__ bias, _Float16* __restrict__ out) {
  int idx = blockIdx.x * blockDim.x + threadIdx.x;
  if (idx >= Bn * 16 * S2) return;
  int x = idx % W2, y = (idx / W2) % H2, o = (idx / S2) % 16, b = idx / (16 * S2);
  float acc = bias[o];
  for (int ic = 0; ic < 4; ++ic) {
    const float* ip = in + (size_t)(b * 4 + ic) * S2;
    for (int ky = 0; ky < 3; ++ky) {
      int yy = y + ky - 1; if (yy < 0 || yy >= H2) continue;
      for (int kx = 0; kx < 3; ++kx) {
        int xx = x + kx - 1; if (xx < 0 || xx >= W2) continue;
        acc += ip[yy * W2 + xx] * w[((o * 4 + ic) * 3 + ky) * 3 + kx];
      }
    }
  }
  out[idx] = (_Float16)fmaxf(acc, 0.f);
}
__global__ void k_conv1_16_16(const _Float16* __restrict__ in, const float* __restrict__ w,
                              const float* __restrict__ bias, _Float16* __restrict__ out) {
  int idx = blockIdx.x * blockDim.x + threadIdx.x;
  if (idx >= Bn * 16 * S2) return;
  int yx = idx % S2, o = (idx / S2) % 16, b = idx / (16 * S2);
  float acc = bias[o];
  for (int ic = 0; ic < 16; ++ic)
    acc += (float)in[(size_t)(b * 16 + ic) * S2 + yx] * w[o * 16 + ic];
  out[idx] = (_Float16)acc;
}
__global__ void k_conv3_16_2(const _Float16* __restrict__ in, const float* __restrict__ w,
                             const float* __restrict__ bias, float* __restrict__ out) {
  int idx = blockIdx.x * blockDim.x + threadIdx.x;
  if (idx >= Bn * 2 * S2) return;
  int x = idx % W2, y = (idx / W2) % H2, o = (idx / S2) % 2, b = idx / (2 * S2);
  float acc = bias[o];
  for (int ic = 0; ic < 16; ++ic) {
    const _Float16* ip = in + (size_t)(b * 16 + ic) * S2;
    for (int ky = 0; ky < 3; ++ky) {
      int yy = y + ky - 1; if (yy < 0 || yy >= H2) continue;
      for (int kx = 0; kx < 3; ++kx) {
        int xx = x + kx - 1; if (xx < 0 || xx >= W2) continue;
        acc += (float)ip[yy * W2 + xx] * w[((o * 16 + ic) * 3 + ky) * 3 + kx];
      }
    }
  }
  out[idx] = acc;
}
__global__ void k_cert_final(const float* __restrict__ ch, const float* __restrict__ certOld,
                             const float* __restrict__ stats, float* __restrict__ outCert) {
  int idx = blockIdx.x * blockDim.x + threadIdx.x;
  if (idx >= Bn * 2 * S2) return;
  int c = (idx / S2) % 2;
  const float cnt = (float)(Bn * S2);
  float m = stats[8 + c] / cnt;
  float var = stats[10 + c] / cnt - m * m;
  outCert[idx] = (ch[idx] - m) * rsqrtf(var + 1e-5f) + certOld[idx];
}

// =================================================================================
extern "C" void kernel_launch(void* const* d_in, const int* in_sizes, int n_in,
                              void* d_out, int out_size, void* d_ws, size_t ws_size,
                              hipStream_t stream) {
  const float* flow   = (const float*)d_in[0];
  const float* cert   = (const float*)d_in[1];
  const float* feat   = (const float*)d_in[2];
  const float* ctx    = (const float*)d_in[3];
  const float* pe_w1  = (const float*)d_in[4];
  const float* pe_b1  = (const float*)d_in[5];
  const float* pe_w2  = (const float*)d_in[6];
  const float* pe_b2  = (const float*)d_in[7];
  const float* pe_w3  = (const float*)d_in[8];
  const float* pe_b3  = (const float*)d_in[9];
  const float* comp_w = (const float*)d_in[10];
  const float* comp_b = (const float*)d_in[11];
  const float* wqk_w  = (const float*)d_in[12];
  const float* wqk_b  = (const float*)d_in[13];
  const float* se_w1  = (const float*)d_in[14];
  const float* se_b1  = (const float*)d_in[15];
  const float* se_w2  = (const float*)d_in[16];
  const float* se_b2  = (const float*)d_in[17];
  const float* cc_w   = (const float*)d_in[18];
  const float* cc_b   = (const float*)d_in[19];
  const float* fus_w  = (const float*)d_in[20];
  const float* fus_b  = (const float*)d_in[21];
  const float* cd_w1  = (const float*)d_in[22];
  const float* cd_b1  = (const float*)d_in[23];
  const float* cd_w2  = (const float*)d_in[24];
  const float* cd_b2  = (const float*)d_in[25];
  const float* cd_w3  = (const float*)d_in[26];
  const float* cd_b3  = (const float*)d_in[27];

  float* outFlow = (float*)d_out;                       // (2,2,512,512)
  float* outCert = outFlow + (size_t)Bn * 2 * S2;       // (2,2,512,512)
  float* outCtx  = outCert + (size_t)Bn * 2 * S2;       // (2,128,256,256)

  // ---- workspace layout (lifetime-aliased, ~345 MB) ----
  char* ws = (char*)d_ws;
  size_t off = 0;
  auto alloc = [&](size_t bytes) -> char* {
    char* p = ws + off;
    off = (off + bytes + 255) & ~(size_t)255;
    return p;
  };
  char* regA = alloc((size_t)NPIX * 320 * 2);           // actAll | fusIn | ch1+ch2
  _Float16* actAll = (_Float16*)regA;
  _Float16* fusIn  = (_Float16*)regA;
  _Float16* ch1    = (_Float16*)regA;
  _Float16* ch2    = (_Float16*)(regA + (size_t)Bn * 16 * S2 * 2);
  char* regB = alloc((size_t)NPIX * 64 * 4);            // qk | cdin+certOld+chout
  float* qkbuf   = (float*)regB;
  float* cdin    = (float*)regB;
  float* certOld = (float*)(regB + (size_t)Bn * 4 * S2 * 4);
  float* chout   = (float*)(regB + (size_t)Bn * 4 * S2 * 4 + (size_t)Bn * 2 * S2 * 4);
  float*    qp     = (float*)alloc((size_t)Bn * PH * PW * 32 * 4);
  float*    kp     = (float*)alloc((size_t)Bn * PH * PW * 32 * 4);
  _Float16* fout   = (_Float16*)alloc((size_t)NPIX * 160 * 2);
  _Float16* ctxNew = (_Float16*)alloc((size_t)NPIX * 192 * 2);
  _Float16* shbuf  = (_Float16*)alloc((size_t)NPIX * 160 * 2);
  float*    sbuf   = (float*)alloc((size_t)NPIX * 48 * 4);
  _Float16* cnew   = (_Float16*)alloc((size_t)NPIX * 128 * 2);
  _Float16* wComp  = (_Float16*)alloc(160 * 320 * 2);
  _Float16* wQk    = (_Float16*)alloc(64 * 160 * 2);
  _Float16* wSe1   = (_Float16*)alloc(160 * 192 * 2);
  _Float16* wSe2   = (_Float16*)alloc(48 * 160 * 2);
  _Float16* wCc    = (_Float16*)alloc(128 * 192 * 2);
  _Float16* wFus   = (_Float16*)alloc(128 * 288 * 2);
  float* biasAll   = (float*)alloc(688 * 4);
  float* bComp = biasAll, *bQk = biasAll + 160, *bSe1 = biasAll + 224;
  float* bSe2 = biasAll + 384, *bCc = biasAll + 432, *bFus = biasAll + 560;
  float* stats = (float*)alloc(16 * 4);

  auto cdiv = [](int a, int b) { return (a + b - 1) / b; };
  const int T = 256;

  // ---- pack weights to f16 [CoPad][Kpad] (zero-padded) ----
  k_pack_weight<<<cdiv(160 * 320, T), T, 0, stream>>>(comp_w, wComp, 160, 290, 160, 320);
  k_pack_weight<<<cdiv(64 * 160, T), T, 0, stream>>>(wqk_w, wQk, 64, 144, 64, 160);
  k_pack_weight<<<cdiv(160 * 192, T), T, 0, stream>>>(se_w1, wSe1, 144, 169, 160, 192);
  k_pack_weight<<<cdiv(48 * 160, T), T, 0, stream>>>(se_w2, wSe2, 48, 144, 48, 160);
  k_pack_weight<<<cdiv(128 * 192, T), T, 0, stream>>>(cc_w, wCc, 128, 169, 128, 192);
  k_pack_weight<<<cdiv(128 * 288, T), T, 0, stream>>>(fus_w, wFus, 128, 272, 128, 288);
  k_pack_bias<<<1, 160, 0, stream>>>(comp_b, bComp, 160, 160);
  k_pack_bias<<<1, 64, 0, stream>>>(wqk_b, bQk, 64, 64);
  k_pack_bias<<<1, 160, 0, stream>>>(se_b1, bSe1, 144, 160);
  k_pack_bias<<<1, 48, 0, stream>>>(se_b2, bSe2, 48, 48);
  k_pack_bias<<<1, 128, 0, stream>>>(cc_b, bCc, 128, 128);
  k_pack_bias<<<1, 128, 0, stream>>>(fus_b, bFus, 128, 128);

  // ---- ctx_all (pe MLP fused) ----
  k_pack_act<<<cdiv(NPIX, T), T, 0, stream>>>(flow, cert, feat, ctx, pe_w1, pe_b1,
                                              pe_w2, pe_b2, pe_w3, pe_b3, actAll);

  dim3 gblk(32, 4);
  // comp: 290->160, relu, f16 out
  k_gemm<<<dim3(NPIX / 16, cdiv(10, 4)), gblk, 0, stream>>>(actAll, wComp, bComp, fout,
                                                            320, 10, 160, 1, 0, 0);
  // wqk: f[:, :144] -> 64, f32 out
  k_gemm<<<dim3(NPIX / 16, 1), gblk, 0, stream>>>(fout, wQk, bQk, qkbuf,
                                                  160, 4, 64, 0, 1, 0);
  // q/k normalize + pad(-1)
  k_qk_prepare<<<cdiv(Bn * PH * PW, T), T, 0, stream>>>(qkbuf, qp, kp);
  // ctx_new = [feat2, 5x5 correlation]
  k_ctx_new<<<cdiv(NPIX, T), T, 0, stream>>>(fout, qp, kp, ctxNew);
  // se1: 169->144 relu (padded to 160)
  k_gemm<<<dim3(NPIX / 16, cdiv(10, 4)), gblk, 0, stream>>>(ctxNew, wSe1, bSe1, shbuf,
                                                            192, 10, 160, 1, 0, 0);
  // se2: 144->48, f32 out
  k_gemm<<<dim3(NPIX / 16, 1), gblk, 0, stream>>>(shbuf, wSe2, bSe2, sbuf,
                                                  160, 3, 48, 0, 1, 0);
  // cc: 169->128, f16 out
  k_gemm<<<dim3(NPIX / 16, 2), gblk, 0, stream>>>(ctxNew, wCc, bCc, cnew,
                                                  192, 8, 128, 0, 0, 0);
  // fus input pack + fus: 272->128, NCHW f32 straight into d_out
  k_fus_in<<<cdiv(NPIX, T), T, 0, stream>>>(cnew, ctx, fusIn);
  k_gemm<<<dim3(NPIX / 16, 2), gblk, 0, stream>>>(fusIn, wFus, bFus, outCtx,
                                                  288, 8, 128, 0, 2, 128);

  // ---- batch-norm stats + convex upsample -> flow_out ----
  k_zero_stats<<<1, 32, 0, stream>>>(stats);
  k_score_stats<<<cdiv(NPIX, T), T, 0, stream>>>(sbuf, stats);
  k_upsample<<<cdiv(NPIX, T), T, 0, stream>>>(sbuf, flow, stats, outFlow);

  // ---- certainty decoder -> cert_out ----
  k_cd_in<<<cdiv(Bn * S2, T), T, 0, stream>>>(sbuf, cert, cdin, certOld);
  k_conv3_4_16<<<cdiv(Bn * 16 * S2, T), T, 0, stream>>>(cdin, cd_w1, cd_b1, ch1);
  k_conv1_16_16<<<cdiv(Bn * 16 * S2, T), T, 0, stream>>>(ch1, cd_w2, cd_b2, ch2);
  k_conv3_16_2<<<cdiv(Bn * 2 * S2, T), T, 0, stream>>>(ch2, cd_w3, cd_b3, chout);
  k_cd_stats<<<cdiv(Bn * 2 * S2, T), T, 0, stream>>>(chout, stats);
  k_cert_final<<<cdiv(Bn * 2 * S2, T), T, 0, stream>>>(chout, certOld, stats, outCert);

  (void)in_sizes; (void)n_in; (void)out_size; (void)ws_size;
}
